// DrugNet_78237124263948
// MI455X (gfx1250) — matrix operand order repaired and checked by reading, not compiled, not run on hardware
//
#include <hip/hip_runtime.h>
#include <hip/hip_bf16.h>

// Problem constants (fixed by the reference)
#define NN 50000
#define EE 600000
#define GG 256
#define DN 128
#define LL 5
#define VV 128
#define BN_EPS 1e-5f

typedef __attribute__((ext_vector_type(16))) __bf16 v16bf;
typedef __attribute__((ext_vector_type(8)))  float v8f;
typedef __attribute__((ext_vector_type(4)))  unsigned int v4u;
typedef __attribute__((ext_vector_type(8)))  int v8i;
typedef __attribute__((ext_vector_type(4)))  int v4i;
typedef __hip_bfloat16 bf16;

// ---------------- utility ----------------
__global__ void k_zero(float* __restrict__ p, int n) {
    int i = blockIdx.x * blockDim.x + threadIdx.x;
    if (i < n) p[i] = 0.0f;
}

// ---------------- AtomEncoder: h[n,c] = sum_f atom_emb[f, x[n,f], c]; h_sum = h ----------------
__global__ void k_atom(const int* __restrict__ x, const float* __restrict__ emb,
                       float* __restrict__ h, float* __restrict__ hsum) {
    int idx = blockIdx.x * blockDim.x + threadIdx.x;
    if (idx >= NN * DN) return;
    int n = idx >> 7, c = idx & 127;
    float s = 0.0f;
#pragma unroll
    for (int f = 0; f < 9; ++f) {
        int v = x[n * 9 + f];
        s += emb[((size_t)(f * VV + v)) * DN + c];
    }
    h[idx] = s;
    hsum[idx] = s;
}

// ---------------- per-edge: msg = relu(h[src] + bond lookup); atomic scatter-add to agg[dst] ----------------
__global__ void k_edge(const int* __restrict__ ei, const int* __restrict__ ea,
                       const float* __restrict__ bemb, const float* __restrict__ h,
                       float* __restrict__ agg) {
    int gid = blockIdx.x * blockDim.x + threadIdx.x;
    int e = gid >> 5, lane = gid & 31;
    if (e >= EE) return;
    int src = ei[e];
    int dst = ei[EE + e];
    int a0 = ea[e * 3 + 0], a1 = ea[e * 3 + 1], a2 = ea[e * 3 + 2];
    int c = lane * 4;
    float4 e0 = *(const float4*)(bemb + ((size_t)(0 * VV + a0)) * DN + c);
    float4 e1 = *(const float4*)(bemb + ((size_t)(1 * VV + a1)) * DN + c);
    float4 e2 = *(const float4*)(bemb + ((size_t)(2 * VV + a2)) * DN + c);
    float4 hv = *(const float4*)(h + (size_t)src * DN + c);
    float m0 = fmaxf(hv.x + e0.x + e1.x + e2.x, 0.0f);
    float m1 = fmaxf(hv.y + e0.y + e1.y + e2.y, 0.0f);
    float m2 = fmaxf(hv.z + e0.z + e1.z + e2.z, 0.0f);
    float m3 = fmaxf(hv.w + e0.w + e1.w + e2.w, 0.0f);
    float* ap = agg + (size_t)dst * DN + c;
    atomicAdd(ap + 0, m0);
    atomicAdd(ap + 1, m1);
    atomicAdd(ap + 2, m2);
    atomicAdd(ap + 3, m3);
}

// ---------------- hin = (1+eps[l])*h + agg, cast to bf16 ----------------
__global__ void k_pre(const float* __restrict__ h, const float* __restrict__ agg,
                      const float* __restrict__ eps, int l, bf16* __restrict__ out) {
    int idx = blockIdx.x * blockDim.x + threadIdx.x;
    if (idx >= NN * DN) return;
    float s = (1.0f + eps[l]) * h[idx] + agg[idx];
    out[idx] = __float2bfloat16(s);
}

// ---------------- column mean/var partials ----------------
__global__ void k_colstats(const float* __restrict__ Z, float* __restrict__ s1,
                           float* __restrict__ s2, int M, int C) {
    int c = threadIdx.x;
    float a = 0.0f, b = 0.0f;
    for (int r = blockIdx.x; r < M; r += gridDim.x) {
        float v = Z[(size_t)r * C + c];
        a += v;
        b += v * v;
    }
    atomicAdd(&s1[c], a);
    atomicAdd(&s2[c], b);
}

// ---------------- BN + relu + bf16 cast (inner MLP hidden) ----------------
__global__ void k_bn1(const float* __restrict__ Z, const float* __restrict__ s1,
                      const float* __restrict__ s2, const float* __restrict__ g,
                      const float* __restrict__ be, bf16* __restrict__ out, int M, int C) {
    int idx = blockIdx.x * blockDim.x + threadIdx.x;
    if (idx >= M * C) return;
    int c = idx % C;
    float m = s1[c] / (float)M;
    float var = s2[c] / (float)M - m * m;
    float v = (Z[idx] - m) * rsqrtf(var + BN_EPS) * g[c] + be[c];
    out[idx] = __float2bfloat16(fmaxf(v, 0.0f));
}

// ---------------- layer-output BN (+optional relu), update h and h_sum ----------------
__global__ void k_bn2(const float* __restrict__ H, const float* __restrict__ s1,
                      const float* __restrict__ s2, const float* __restrict__ g,
                      const float* __restrict__ b, float* __restrict__ h,
                      float* __restrict__ hsum, int do_relu) {
    int idx = blockIdx.x * blockDim.x + threadIdx.x;
    if (idx >= NN * DN) return;
    int c = idx & 127;
    float m = s1[c] / (float)NN;
    float var = s2[c] / (float)NN - m * m;
    float v = (H[idx] - m) * rsqrtf(var + BN_EPS) * g[c] + b[c];
    if (do_relu) v = fmaxf(v, 0.0f);
    h[idx] = v;
    hsum[idx] += v;
}

// ---------------- global mean pool (scatter) ----------------
__global__ void k_pool(const float* __restrict__ hsum, const int* __restrict__ batch,
                       float* __restrict__ rsum, float* __restrict__ cnt) {
    int idx = blockIdx.x * blockDim.x + threadIdx.x;
    if (idx >= NN * DN) return;
    int n = idx >> 7, c = idx & 127;
    int g = batch[n];
    atomicAdd(&rsum[(size_t)g * DN + c], hsum[idx]);
    if (c == 0) atomicAdd(&cnt[g], 1.0f);
}

__global__ void k_rep(const float* __restrict__ rsum, const float* __restrict__ cnt,
                      bf16* __restrict__ rep) {
    int idx = blockIdx.x * blockDim.x + threadIdx.x;
    if (idx >= GG * DN) return;
    int g = idx >> 7;
    rep[idx] = __float2bfloat16(rsum[idx] / fmaxf(cnt[g], 1.0f));
}

// ---------------- pack f32 weight [K,C] row-major into wave32 bf16 B-fragment layout ----------------
// layout: [K/32][C/16][lane(32)][elem(16)] ; col = ct*16 + lane%16 ; k = ks*32 + elem + (lane/16)*16
__global__ void k_pack(const float* __restrict__ W, bf16* __restrict__ Bp, int K, int C) {
    int idx = blockIdx.x * blockDim.x + threadIdx.x;
    if (idx >= K * C) return;
    int i = idx & 15;
    int lane = (idx >> 4) & 31;
    int rest = idx >> 9;
    int CT = C >> 4;
    int ct = rest % CT;
    int ks = rest / CT;
    int col = ct * 16 + (lane & 15);
    int k = ks * 32 + i + (lane >> 4) * 16;
    Bp[idx] = __float2bfloat16(W[(size_t)k * C + col]);
}

// ---------------- TDM: stage a 2D panel of packed-B into LDS ----------------
// panel: `rows` rows of 4096 bytes, global row stride = CT*1024 bytes (D# in 8-byte units)
__device__ inline void tdm_load_b_panel(const void* gptr, unsigned lds_off,
                                        unsigned rows, unsigned strideUnits) {
    unsigned long long ga = (unsigned long long)(uintptr_t)gptr;
    const unsigned dim0 = 512u;   // 4096B / 8B units, also tile_dim0
    v4u g0;
    g0[0] = 1u;                                        // count=1, user descriptor
    g0[1] = lds_off;                                   // lds_addr (bytes)
    g0[2] = (unsigned)(ga & 0xFFFFFFFFu);              // global_addr[31:0]
    g0[3] = (unsigned)((ga >> 32) & 0x01FFFFFFu)       // global_addr[56:32]
          | 0x80000000u;                               // type=2 ("image")
    v8i g1;
    g1[0] = (int)(3u << 16);                           // data_size=8B, no multicast/pad/iter
    g1[1] = (int)((dim0 & 0xFFFFu) << 16);             // tensor_dim0[15:0]
    g1[2] = (int)((dim0 >> 16) | ((rows & 0xFFFFu) << 16)); // dim0[31:16] | tensor_dim1[15:0]
    g1[3] = (int)((rows >> 16) | (dim0 << 16));        // dim1[31:16] | tile_dim0
    g1[4] = (int)(rows & 0xFFFFu);                     // tile_dim1 (tile_dim2=0)
    g1[5] = (int)strideUnits;                          // tensor_dim0_stride[31:0]
    g1[6] = 0;                                         // stride hi / dim1_stride (unused, 2D)
    g1[7] = 0;
    v4i z4 = {0, 0, 0, 0};
    v8i z8 = {0, 0, 0, 0, 0, 0, 0, 0};
    // clang-23 / therock-10.0 lane: 6-arg form
    __builtin_amdgcn_tensor_load_to_lds(g0, g1, z4, z4, z8, 0);
}

// ---------------- WMMA bf16 GEMM: C[M,Cc] = A[M,K] @ B + bias ----------------
// block = 128 threads = 4 waves; each wave: one 16-row tile x 64 cols (4 accumulators);
// all 4 waves share a TDM-staged 32KB B panel in LDS. grid = (ceil(M/64), Cc/64)
__global__ __launch_bounds__(128) void k_gemm(const bf16* __restrict__ A,
                                              const bf16* __restrict__ Bp,
                                              const float* __restrict__ bias,
                                              float* __restrict__ Cf, bf16* __restrict__ Cbf,
                                              int M, int K, int Cc, int do_relu) {
    __shared__ __align__(16) unsigned char sB[32768]; // 8 K-steps x 4KB (64 cols worth)
    const int wave = threadIdx.x >> 5;
    const int lane = threadIdx.x & 31;
    const int Mt = M >> 4;
    const int rowTile = blockIdx.x * 4 + wave;
    const bool active = rowTile < Mt;           // wave-uniform (EXEC stays all-ones for WMMA)
    const int cg = blockIdx.y;                  // 64-column group
    const int CT = Cc >> 4;
    const int grp = lane >> 4, lr = lane & 15;
    const int row0 = rowTile << 4;
    v8f acc[4] = {v8f{}, v8f{}, v8f{}, v8f{}};

    for (int kb = 0; kb < K; kb += 256) {
        const int rows = (K - kb >= 256) ? 8 : ((K - kb) >> 5);
        if (wave == 0) {
            const void* gp = (const char*)Bp + ((size_t)(kb >> 5) * CT + cg * 4) * 1024;
            tdm_load_b_panel(gp, (unsigned)(uintptr_t)&sB[0], (unsigned)rows,
                             (unsigned)(CT * 128));
            __builtin_amdgcn_s_wait_tensorcnt(0);
        }
        __syncthreads();
        if (active) {
            for (int r = 0; r < rows; ++r) {
                const int ks = kb + (r << 5);
                union { uint4 u[2]; v16bf v; } af;
                const bf16* ap = A + (size_t)(row0 + lr) * K + ks + grp * 16;
                af.u[0] = *(const uint4*)(ap);
                af.u[1] = *(const uint4*)(ap + 8);
                if (ks + 32 < K) __builtin_prefetch(ap + 32, 0, 1); // global_prefetch_b8
#pragma unroll
                for (int ctl = 0; ctl < 4; ++ctl) {
                    union { uint4 u[2]; v16bf v; } bfr;
                    const uint4* sp =
                        (const uint4*)(&sB[0] + r * 4096 + ctl * 1024 + lane * 32);
                    bfr.u[0] = sp[0];                 // ds_load_b128
                    bfr.u[1] = sp[1];
                    acc[ctl] = __builtin_amdgcn_wmma_f32_16x16x32_bf16(
                        false, af.v, false, bfr.v, (short)0, acc[ctl], false, false);
                }
            }
        }
        __syncthreads();
    }

    if (active) {
#pragma unroll
        for (int ctl = 0; ctl < 4; ++ctl) {
            const int col = (cg * 4 + ctl) * 16 + lr;
            const float bv = bias[col];
#pragma unroll
            for (int r = 0; r < 8; ++r) {
                int row = row0 + r + grp * 8; // D layout: VGPR r -> M=r / M=r+8
                float v = acc[ctl][r] + bv;
                if (do_relu) v = fmaxf(v, 0.0f);
                size_t o = (size_t)row * Cc + col;
                if (Cf) Cf[o] = v;
                if (Cbf) Cbf[o] = __float2bfloat16(v);
            }
        }
    }
}

// ---------------- final 512 -> 1 projection per graph ----------------
__global__ void k_fc4(const bf16* __restrict__ a, const float* __restrict__ w,
                      const float* __restrict__ b, float* __restrict__ out) {
    int g = blockIdx.x * blockDim.x + threadIdx.x;
    if (g >= GG) return;
    float s = b[0];
    for (int k = 0; k < 512; ++k)
        s += __bfloat162float(a[(size_t)g * 512 + k]) * w[k];
    out[g] = s;
}

extern "C" void kernel_launch(void* const* d_in, const int* in_sizes, int n_in,
                              void* d_out, int out_size, void* d_ws, size_t ws_size,
                              hipStream_t stream) {
    (void)in_sizes; (void)n_in; (void)out_size; (void)ws_size;
    const int*   x        = (const int*)d_in[0];
    const int*   ei       = (const int*)d_in[1];
    const int*   ea       = (const int*)d_in[2];
    const int*   batch    = (const int*)d_in[3];
    const float* atom_emb = (const float*)d_in[4];
    const float* bond_emb = (const float*)d_in[5];
    const float* eps      = (const float*)d_in[6];
    const float* W1       = (const float*)d_in[7];
    const float* b1       = (const float*)d_in[8];
    const float* g1       = (const float*)d_in[9];
    const float* be1      = (const float*)d_in[10];
    const float* W2       = (const float*)d_in[11];
    const float* b2       = (const float*)d_in[12];
    const float* bn_g     = (const float*)d_in[13];
    const float* bn_b     = (const float*)d_in[14];
    const float* fcW1     = (const float*)d_in[15];
    const float* fcb1     = (const float*)d_in[16];
    const float* fcW2     = (const float*)d_in[17];
    const float* fcb2     = (const float*)d_in[18];
    const float* fcW3     = (const float*)d_in[19];
    const float* fcb3     = (const float*)d_in[20];
    const float* fcW4     = (const float*)d_in[21];
    const float* fcb4     = (const float*)d_in[22];
    float*       out      = (float*)d_out;

    // -------- workspace layout (256B aligned) --------
    char* ws = (char*)d_ws;
    size_t off = 0;
    auto alloc = [&](size_t bytes) {
        size_t r = off;
        off = (off + bytes + 255) & ~(size_t)255;
        return r;
    };
    float* h     = (float*)(ws + alloc((size_t)NN * DN * 4));
    float* hsum  = (float*)(ws + alloc((size_t)NN * DN * 4));
    float* agg   = (float*)(ws + alloc((size_t)NN * DN * 4));
    bf16*  hinbf = (bf16*) (ws + alloc((size_t)NN * DN * 2));
    float* Z     = (float*)(ws + alloc((size_t)NN * 256 * 4));
    bf16*  Zbf   = (bf16*) (ws + alloc((size_t)NN * 256 * 2));
    float* H2    = (float*)(ws + alloc((size_t)NN * DN * 4));
    float* s1a   = (float*)(ws + alloc(256 * 4));
    float* s2a   = (float*)(ws + alloc(256 * 4));
    float* s1b   = (float*)(ws + alloc(128 * 4));
    float* s2b   = (float*)(ws + alloc(128 * 4));
    bf16*  W1p   = (bf16*) (ws + alloc((size_t)128 * 256 * 2));
    bf16*  W2p   = (bf16*) (ws + alloc((size_t)256 * 128 * 2));
    bf16*  f1p   = (bf16*) (ws + alloc((size_t)128 * 1024 * 2));
    bf16*  f2p   = (bf16*) (ws + alloc((size_t)1024 * 1024 * 2));
    bf16*  f3p   = (bf16*) (ws + alloc((size_t)1024 * 512 * 2));
    float* rsum  = (float*)(ws + alloc((size_t)GG * DN * 4));
    float* cnt   = (float*)(ws + alloc(GG * 4));
    bf16*  repbf = (bf16*) (ws + alloc((size_t)GG * DN * 2));
    bf16*  a1bf  = (bf16*) (ws + alloc((size_t)GG * 1024 * 2));
    bf16*  a2bf  = (bf16*) (ws + alloc((size_t)GG * 1024 * 2));
    bf16*  a3bf  = (bf16*) (ws + alloc((size_t)GG * 512 * 2));

    const int TPB = 256;
    const int nd_blocks = (NN * DN + TPB - 1) / TPB;        // 25000
    const int z_blocks  = (NN * 256 + TPB - 1) / TPB;       // 50000
    const int e_blocks  = (EE * 32 + TPB - 1) / TPB;        // 75000
    const int gx_big    = (NN / 16 + 3) / 4;                // 782 (3125 row tiles, 4/block)
    const int gx_fc     = (GG / 16 + 3) / 4;                // 4

    // -------- pack weights into B-fragment layout (once per launch) --------
    k_pack<<<(128 * 256 + TPB - 1) / TPB, TPB, 0, stream>>>(W1, W1p, 128, 256);
    k_pack<<<(256 * 128 + TPB - 1) / TPB, TPB, 0, stream>>>(W2, W2p, 256, 128);
    k_pack<<<(128 * 1024 + TPB - 1) / TPB, TPB, 0, stream>>>(fcW1, f1p, 128, 1024);
    k_pack<<<(1024 * 1024 + TPB - 1) / TPB, TPB, 0, stream>>>(fcW2, f2p, 1024, 1024);
    k_pack<<<(1024 * 512 + TPB - 1) / TPB, TPB, 0, stream>>>(fcW3, f3p, 1024, 512);

    // -------- atom encoder --------
    k_atom<<<nd_blocks, TPB, 0, stream>>>(x, atom_emb, h, hsum);

    // -------- GIN layers --------
    for (int l = 0; l < LL; ++l) {
        k_zero<<<nd_blocks, TPB, 0, stream>>>(agg, NN * DN);
        k_edge<<<e_blocks, TPB, 0, stream>>>(ei, ea, bond_emb + (size_t)l * 3 * VV * DN, h, agg);
        k_pre<<<nd_blocks, TPB, 0, stream>>>(h, agg, eps, l, hinbf);
        // GEMM1: [N,128] @ [128,256]
        k_gemm<<<dim3(gx_big, 256 / 64), 128, 0, stream>>>(hinbf, W1p, b1, Z, nullptr,
                                                           NN, 128, 256, 0);
        k_zero<<<2, 256, 0, stream>>>(s1a, 256);
        k_zero<<<2, 256, 0, stream>>>(s2a, 256);
        k_colstats<<<512, 256, 0, stream>>>(Z, s1a, s2a, NN, 256);
        k_bn1<<<z_blocks, TPB, 0, stream>>>(Z, s1a, s2a, g1, be1, Zbf, NN, 256);
        // GEMM2: [N,256] @ [256,128]
        k_gemm<<<dim3(gx_big, 128 / 64), 128, 0, stream>>>(Zbf, W2p, b2, H2, nullptr,
                                                           NN, 256, 128, 0);
        k_zero<<<1, 256, 0, stream>>>(s1b, 128);
        k_zero<<<1, 256, 0, stream>>>(s2b, 128);
        k_colstats<<<512, 128, 0, stream>>>(H2, s1b, s2b, NN, 128);
        k_bn2<<<nd_blocks, TPB, 0, stream>>>(H2, s1b, s2b, bn_g + (size_t)l * DN,
                                             bn_b + (size_t)l * DN, h, hsum,
                                             (l < LL - 1) ? 1 : 0);
    }

    // -------- global mean pool --------
    k_zero<<<(GG * DN + TPB - 1) / TPB, TPB, 0, stream>>>(rsum, GG * DN);
    k_zero<<<1, 256, 0, stream>>>(cnt, GG);
    k_pool<<<nd_blocks, TPB, 0, stream>>>(hsum, batch, rsum, cnt);
    k_rep<<<(GG * DN + TPB - 1) / TPB, TPB, 0, stream>>>(rsum, cnt, repbf);

    // -------- FC head (WMMA) --------
    k_gemm<<<dim3(gx_fc, 1024 / 64), 128, 0, stream>>>(repbf, f1p, fcb1, nullptr, a1bf,
                                                       GG, 128, 1024, 1);
    k_gemm<<<dim3(gx_fc, 1024 / 64), 128, 0, stream>>>(a1bf, f2p, fcb2, nullptr, a2bf,
                                                       GG, 1024, 1024, 1);
    k_gemm<<<dim3(gx_fc, 512 / 64), 128, 0, stream>>>(a2bf, f3p, fcb3, nullptr, a3bf,
                                                      GG, 1024, 512, 1);
    k_fc4<<<1, 256, 0, stream>>>(a3bf, fcW4, fcb4, out);
}
